// GNNEncoder_24146306138777
// MI455X (gfx1250) — matrix operand order repaired and checked by reading, not compiled, not run on hardware
//
#include <hip/hip_runtime.h>

typedef __attribute__((ext_vector_type(16))) __bf16 v16bf;
typedef __attribute__((ext_vector_type(8)))  __bf16 v8bf;
typedef __attribute__((ext_vector_type(8)))  float  v8f;

#define N_NODES 50000
#define N_EDGES 800000
#define F_IN    128
#define F_HID   256
#define F_OUT   256
#define N_GRAPH 64
#define M_TOT   (N_EDGES + N_NODES)
#define NEG_ATT 0.2f
#define NEG_ACT 0.01f
#define NTW     8      // n-tiles per wave in the WMMA GEMM

// ---------------- helpers ----------------
__device__ __forceinline__ unsigned fkey(float f) {
    unsigned u = __float_as_uint(f);
    return (u & 0x80000000u) ? ~u : (u | 0x80000000u);
}
__device__ __forceinline__ float fkey_inv(unsigned k) {
    unsigned u = (k & 0x80000000u) ? (k ^ 0x80000000u) : ~k;
    return __uint_as_float(u);
}

// ---------------- conversion / packing ----------------
__global__ void f32_to_bf16_kernel(const float* __restrict__ in,
                                   __bf16* __restrict__ out, int n) {
    int i = blockIdx.x * blockDim.x + threadIdx.x;
    if (i < n) out[i] = (__bf16)in[i];
}

__global__ void leaky_to_bf16_kernel(const float* __restrict__ in,
                                     __bf16* __restrict__ out, int n) {
    int i = blockIdx.x * blockDim.x + threadIdx.x;
    if (i < n) {
        float v = in[i];
        v = v > 0.0f ? v : NEG_ACT * v;
        out[i] = (__bf16)v;
    }
}

// Pack W[K x Nc] (row major f32) into per-(ktile,ntile) WMMA B fragments:
// fragment element e of lane L holds B[k, n] with n = ntile*16 + (L&15),
// k = ktile*32 + (L<16 ? 0 : 16) + e.
__global__ void pack_w_kernel(const float* __restrict__ W,
                              __bf16* __restrict__ Wp, int K, int Nc) {
    int idx = blockIdx.x * blockDim.x + threadIdx.x;
    if (idx >= K * Nc) return;
    int k = idx / Nc, n = idx % Nc;
    int kt = k >> 5, kk = k & 31;
    int nt = n >> 4, c = n & 15;
    int lane = c + ((kk & 16) ? 16 : 0);
    int e = kk & 15;
    int NT = Nc >> 4;
    Wp[(((size_t)kt * NT + nt) * 32 + lane) * 16 + e] = (__bf16)W[idx];
}

// -------- WMMA GEMM: C[rows x Nc] = A[rows x K] * B, NTW n-tiles / wave ----
__global__ __launch_bounds__(256)
void gemm_bf16_wmma_kernel(const __bf16* __restrict__ A,
                           const __bf16* __restrict__ Bp,
                           float* __restrict__ C,
                           int rows, int K, int Nc) {
    const int lane = threadIdx.x & 31;
    const int wave = blockIdx.x * (blockDim.x >> 5) + (threadIdx.x >> 5);
    const int ntiles = Nc >> 4;
    const int groups = ntiles / NTW;
    const int mt = wave / groups;
    const int g  = wave % groups;
    if (mt * 16 >= rows) return;                 // wave-uniform
    const int nt0 = g * NTW;

    const int r  = mt * 16 + (lane & 15);        // A row for this lane
    const int ko = (lane < 16) ? 0 : 16;         // K sub-offset for A frag
    const __bf16* arow = A + (size_t)r * K;
    const v16bf* bbase = (const v16bf*)Bp;
    const int ktiles = K >> 5;

    v8f acc[NTW];
#pragma unroll
    for (int j = 0; j < NTW; ++j) acc[j] = (v8f){};

    for (int kt = 0; kt < ktiles; ++kt) {
        // A fragment: elems 0..15 = K[kt*32+ko .. +15]
        v8bf lo = *(const v8bf*)(arow + kt * 32 + ko);
        v8bf hi = *(const v8bf*)(arow + kt * 32 + 8 + ko);
        v16bf a;
#pragma unroll
        for (int i = 0; i < 8; ++i) { a[i] = lo[i]; a[i + 8] = hi[i]; }
        const v16bf* brow = bbase + ((size_t)kt * ntiles + nt0) * 32 + lane;
#pragma unroll
        for (int j = 0; j < NTW; ++j) {
            v16bf b = brow[(size_t)j * 32];      // contiguous 32B per lane
            acc[j] = __builtin_amdgcn_wmma_f32_16x16x32_bf16(
                false, a, false, b, (short)0, acc[j], false, false);
        }
    }

    // D layout: elem v -> row M = (lane<16 ? v : v+8), col = lane&15
    const int colb  = lane & 15;
    const int mbase = mt * 16 + ((lane < 16) ? 0 : 8);
#pragma unroll
    for (int j = 0; j < NTW; ++j) {
        int col = (nt0 + j) * 16 + colb;
#pragma unroll
        for (int v = 0; v < 8; ++v)
            C[(size_t)(mbase + v) * Nc + col] = acc[j][v];
    }
}

// ---------------- attention logits per node ----------------
__global__ __launch_bounds__(256)
void alphas_kernel(const float* __restrict__ h,
                   const float* __restrict__ a_s, const float* __restrict__ a_d,
                   float* __restrict__ asrc, float* __restrict__ adst,
                   int n, int D) {
    const int lane = threadIdx.x & 31;
    const int wave = blockIdx.x * (blockDim.x >> 5) + (threadIdx.x >> 5);
    const int nw = gridDim.x * (blockDim.x >> 5);
    for (int i = wave; i < n; i += nw) {
        const float* row = h + (size_t)i * D;
        float ss = 0.0f, sd = 0.0f;
        for (int f = lane; f < D; f += 32) {
            float v = row[f];
            ss += v * a_s[f];
            sd += v * a_d[f];
        }
#pragma unroll
        for (int o = 16; o > 0; o >>= 1) {
            ss += __shfl_xor(ss, o, 32);
            sd += __shfl_xor(sd, o, 32);
        }
        if (lane == 0) { asrc[i] = ss; adst[i] = sd; }
    }
}

// ---------------- init kernels ----------------
__global__ void init_bias_kernel(float* __restrict__ out,
                                 const float* __restrict__ bias, int n, int D) {
    int i = blockIdx.x * blockDim.x + threadIdx.x;
    if (i < n) out[i] = bias[i & (D - 1)];       // D is a power of two
}
__global__ void init_softmax_kernel(unsigned* __restrict__ mkeys,
                                    float* __restrict__ denom, int n) {
    int i = blockIdx.x * blockDim.x + threadIdx.x;
    if (i < n) { mkeys[i] = 0u; denom[i] = 0.0f; }
}
__global__ void zero_kernel(float* __restrict__ p, int n) {
    int i = blockIdx.x * blockDim.x + threadIdx.x;
    if (i < n) p[i] = 0.0f;
}

// ---------------- edge passes ----------------
__global__ void edge_max_kernel(const int* __restrict__ ei,
                                const float* __restrict__ asrc,
                                const float* __restrict__ adst,
                                unsigned* __restrict__ mkeys, int E, int M) {
    int i = blockIdx.x * blockDim.x + threadIdx.x;
    if (i >= M) return;
    int s, d;
    if (i < E) { s = ei[i]; d = ei[E + i]; } else { s = d = i - E; }
    float e = asrc[s] + adst[d];
    e = e > 0.0f ? e : NEG_ATT * e;
    atomicMax(&mkeys[d], fkey(e));
}

__global__ void edge_exp_kernel(const int* __restrict__ ei,
                                const float* __restrict__ asrc,
                                const float* __restrict__ adst,
                                const unsigned* __restrict__ mkeys,
                                float* __restrict__ denom,
                                float* __restrict__ pbuf, int E, int M) {
    int i = blockIdx.x * blockDim.x + threadIdx.x;
    if (i >= M) return;
    int s, d;
    if (i < E) { s = ei[i]; d = ei[E + i]; } else { s = d = i - E; }
    float e = asrc[s] + adst[d];
    e = e > 0.0f ? e : NEG_ATT * e;
    float p = expf(e - fkey_inv(mkeys[d]));
    pbuf[i] = p;
    atomicAdd(&denom[d], p);
}

// one wave per edge; float4 gather of h[src], scalar f32 atomic scatter
__global__ __launch_bounds__(256)
void edge_scatter_kernel(const int* __restrict__ ei,
                         const float* __restrict__ pbuf,
                         const float* __restrict__ denom,
                         const float* __restrict__ h,
                         float* __restrict__ out, int E, int M, int D) {
    const int lane = threadIdx.x & 31;
    const int wave = blockIdx.x * (blockDim.x >> 5) + (threadIdx.x >> 5);
    const int nw = gridDim.x * (blockDim.x >> 5);
    const int D4 = D >> 2;
    for (int i = wave; i < M; i += nw) {
        int s, d;
        if (i < E) { s = ei[i]; d = ei[E + i]; } else { s = d = i - E; }
        float coef = pbuf[i] / denom[d];
        const float4* hs = (const float4*)(h + (size_t)s * D);
        float* od = out + (size_t)d * D;
        for (int f4 = lane; f4 < D4; f4 += 32) {
            float4 v = hs[f4];
            int f = f4 << 2;
            atomicAdd(&od[f + 0], v.x * coef);
            atomicAdd(&od[f + 1], v.y * coef);
            atomicAdd(&od[f + 2], v.z * coef);
            atomicAdd(&od[f + 3], v.w * coef);
        }
    }
}

// ---------------- graph pooling ----------------
__global__ __launch_bounds__(256)
void pool_kernel(const float* __restrict__ node, const int* __restrict__ batch,
                 float* __restrict__ graph, int n, int D) {
    const int lane = threadIdx.x & 31;
    const int wave = blockIdx.x * (blockDim.x >> 5) + (threadIdx.x >> 5);
    const int nw = gridDim.x * (blockDim.x >> 5);
    const int D4 = D >> 2;
    for (int i = wave; i < n; i += nw) {
        int b = batch[i];
        const float4* ni = (const float4*)(node + (size_t)i * D);
        float* g = graph + (size_t)b * D;
        for (int f4 = lane; f4 < D4; f4 += 32) {
            float4 v = ni[f4];
            int f = f4 << 2;
            atomicAdd(&g[f + 0], v.x);
            atomicAdd(&g[f + 1], v.y);
            atomicAdd(&g[f + 2], v.z);
            atomicAdd(&g[f + 3], v.w);
        }
    }
}

// ---------------- host orchestration ----------------
extern "C" void kernel_launch(void* const* d_in, const int* in_sizes, int n_in,
                              void* d_out, int out_size, void* d_ws, size_t ws_size,
                              hipStream_t stream) {
    (void)in_sizes; (void)n_in; (void)out_size; (void)ws_size;

    const float* x    = (const float*)d_in[0];
    const int*   ei   = (const int*)d_in[1];
    const int*   bat  = (const int*)d_in[2];
    const float* W1   = (const float*)d_in[3];
    const float* as1  = (const float*)d_in[4];
    const float* ad1  = (const float*)d_in[5];
    const float* b1   = (const float*)d_in[6];
    const float* W2   = (const float*)d_in[7];
    const float* as2  = (const float*)d_in[8];
    const float* ad2  = (const float*)d_in[9];
    const float* b2   = (const float*)d_in[10];

    float* node_out  = (float*)d_out;                           // [N, 256]
    float* graph_out = (float*)d_out + (size_t)N_NODES * F_OUT; // [64, 256]

    // workspace carve-out (256B aligned)
    char*  ws  = (char*)d_ws;
    size_t off = 0;
    auto take = [&](size_t bytes) -> void* {
        void* p = ws + off;
        off = (off + bytes + 255) & ~(size_t)255;
        return p;
    };
    float*    h_f32  = (float*)   take((size_t)N_NODES * F_HID * 4); // h1, then h2
    float*    agg1   = (float*)   take((size_t)N_NODES * F_HID * 4);
    __bf16*   x_bf   = (__bf16*)  take((size_t)N_NODES * F_IN  * 2);
    __bf16*   ha_bf  = (__bf16*)  take((size_t)N_NODES * F_HID * 2);
    __bf16*   w1p    = (__bf16*)  take((size_t)F_IN  * F_HID * 2);
    __bf16*   w2p    = (__bf16*)  take((size_t)F_HID * F_OUT * 2);
    float*    asrc   = (float*)   take((size_t)N_NODES * 4);
    float*    adst   = (float*)   take((size_t)N_NODES * 4);
    unsigned* mkeys  = (unsigned*)take((size_t)N_NODES * 4);
    float*    denom  = (float*)   take((size_t)N_NODES * 4);
    float*    pbuf   = (float*)   take((size_t)M_TOT * 4);

    const int T = 256;
    auto cdiv = [](long long a, long long b) { return (int)((a + b - 1) / b); };

    // ---- precision prep ----
    f32_to_bf16_kernel<<<cdiv((long long)N_NODES * F_IN, T), T, 0, stream>>>(
        x, x_bf, N_NODES * F_IN);
    pack_w_kernel<<<cdiv(F_IN * F_HID, T), T, 0, stream>>>(W1, w1p, F_IN, F_HID);
    pack_w_kernel<<<cdiv(F_HID * F_OUT, T), T, 0, stream>>>(W2, w2p, F_HID, F_OUT);

    // ======== layer 1 ========
    {
        int waves = (N_NODES / 16) * ((F_HID / 16) / NTW);
        gemm_bf16_wmma_kernel<<<cdiv(waves, 8), T, 0, stream>>>(
            x_bf, w1p, h_f32, N_NODES, F_IN, F_HID);
    }
    alphas_kernel<<<cdiv(N_NODES, 8), T, 0, stream>>>(h_f32, as1, ad1, asrc, adst,
                                                      N_NODES, F_HID);
    init_bias_kernel<<<cdiv((long long)N_NODES * F_HID, T), T, 0, stream>>>(
        agg1, b1, N_NODES * F_HID, F_HID);
    init_softmax_kernel<<<cdiv(N_NODES, T), T, 0, stream>>>(mkeys, denom, N_NODES);
    edge_max_kernel<<<cdiv(M_TOT, T), T, 0, stream>>>(ei, asrc, adst, mkeys,
                                                      N_EDGES, M_TOT);
    edge_exp_kernel<<<cdiv(M_TOT, T), T, 0, stream>>>(ei, asrc, adst, mkeys,
                                                      denom, pbuf, N_EDGES, M_TOT);
    edge_scatter_kernel<<<cdiv(M_TOT, 8), T, 0, stream>>>(ei, pbuf, denom, h_f32,
                                                          agg1, N_EDGES, M_TOT, F_HID);
    // activation + bf16 for next GEMM
    leaky_to_bf16_kernel<<<cdiv((long long)N_NODES * F_HID, T), T, 0, stream>>>(
        agg1, ha_bf, N_NODES * F_HID);

    // ======== layer 2 ========
    {
        int waves = (N_NODES / 16) * ((F_OUT / 16) / NTW);
        gemm_bf16_wmma_kernel<<<cdiv(waves, 8), T, 0, stream>>>(
            ha_bf, w2p, h_f32, N_NODES, F_HID, F_OUT);
    }
    alphas_kernel<<<cdiv(N_NODES, 8), T, 0, stream>>>(h_f32, as2, ad2, asrc, adst,
                                                      N_NODES, F_OUT);
    init_bias_kernel<<<cdiv((long long)N_NODES * F_OUT, T), T, 0, stream>>>(
        node_out, b2, N_NODES * F_OUT, F_OUT);
    init_softmax_kernel<<<cdiv(N_NODES, T), T, 0, stream>>>(mkeys, denom, N_NODES);
    edge_max_kernel<<<cdiv(M_TOT, T), T, 0, stream>>>(ei, asrc, adst, mkeys,
                                                      N_EDGES, M_TOT);
    edge_exp_kernel<<<cdiv(M_TOT, T), T, 0, stream>>>(ei, asrc, adst, mkeys,
                                                      denom, pbuf, N_EDGES, M_TOT);
    edge_scatter_kernel<<<cdiv(M_TOT, 8), T, 0, stream>>>(ei, pbuf, denom, h_f32,
                                                          node_out, N_EDGES, M_TOT, F_OUT);

    // ======== graph pooling ========
    zero_kernel<<<cdiv(N_GRAPH * F_OUT, T), T, 0, stream>>>(graph_out, N_GRAPH * F_OUT);
    pool_kernel<<<cdiv(N_NODES, 8), T, 0, stream>>>(node_out, bat, graph_out,
                                                    N_NODES, F_OUT);
}
